// GraphCNN_73212012528326
// MI455X (gfx1250) — matrix operand order repaired and checked by reading, not compiled, not run on hardware
//
#include <hip/hip_runtime.h>
#include <hip/hip_bf16.h>
#include <math.h>

#define NN 2048
#define QQ 9
#define DD 256
#define LL 4
#define BB 32
#define EE 32768
#define ROWS (BB * NN) /* 65536 */

typedef __attribute__((ext_vector_type(16))) __bf16 v16bf;
typedef __attribute__((ext_vector_type(8)))  float  v8f;
typedef __attribute__((ext_vector_type(4)))  unsigned int v4u;

union Vec16 { v16bf h; v4u q2[2]; };

__device__ __forceinline__ float u2f(unsigned int u) {
    union { unsigned int u; float f; } c; c.u = u; return c.f;
}
__device__ __forceinline__ unsigned int f2u(float f) {
    union { unsigned int u; float f; } c; c.f = f; return c.u;
}
__device__ __forceinline__ float bflo(unsigned int w) { return u2f(w << 16); }
__device__ __forceinline__ float bfhi(unsigned int w) { return u2f(w & 0xFFFF0000u); }
__device__ __forceinline__ float bf2f(unsigned short s) { return u2f(((unsigned int)s) << 16); }
__device__ __forceinline__ unsigned short f2bf(float f) {
    unsigned int x = f2u(f);
    unsigned int r = x + 0x7FFFu + ((x >> 16) & 1u);   // round to nearest even
    return (unsigned short)(r >> 16);
}

// ---------------------------------------------------------------------------
// CSR build (deterministic, no atomics — bit-identical across graph replays)
// ---------------------------------------------------------------------------
__global__ void csr_count(const int* __restrict__ src, int* __restrict__ counts) {
    int n = blockIdx.x * 256 + threadIdx.x;
    if (n >= NN) return;
    int c = 0;
    for (int e = 0; e < EE; ++e) c += (src[e] == n) ? 1 : 0;
    counts[n] = c;
}

__global__ void csr_scan(const int* __restrict__ counts, int* __restrict__ rowPtr) {
    if (threadIdx.x == 0 && blockIdx.x == 0) {
        int acc = 0;
        for (int n = 0; n < NN; ++n) { rowPtr[n] = acc; acc += counts[n]; }
        rowPtr[NN] = acc;
    }
}

__global__ void csr_fill(const int* __restrict__ src, const int* __restrict__ dst,
                         const int* __restrict__ rowPtr, int* __restrict__ cols) {
    int n = blockIdx.x * 256 + threadIdx.x;
    if (n >= NN) return;
    int p = rowPtr[n];
    for (int e = 0; e < EE; ++e)
        if (src[e] == n) cols[p++] = dst[e];
}

// ---------------------------------------------------------------------------
// h0 = batch_feature with rows [0,Q) replaced by emotion_emb; stored bf16
// ---------------------------------------------------------------------------
__global__ void __launch_bounds__(256)
init_h(const float* __restrict__ bf, const float* __restrict__ emo,
       unsigned short* __restrict__ H) {
    size_t idx = (size_t)blockIdx.x * 256 + threadIdx.x;   // < ROWS*DD
    int d = (int)(idx & (DD - 1));
    int n = (int)((idx >> 8) & (NN - 1));
    float v = (n < QQ) ? emo[n * DD + d] : bf[idx];
    H[idx] = f2bf(v);
}

// Transpose-convert weights: Wt[l][n][k] = W[l][k][n], f32 -> bf16
__global__ void __launch_bounds__(256)
conv_w(const float* __restrict__ w1, const float* __restrict__ w2,
       unsigned short* __restrict__ wt1, unsigned short* __restrict__ wt2) {
    int idx = blockIdx.x * 256 + threadIdx.x;   // < LL*DD*DD
    int k = idx & 255;
    int n = (idx >> 8) & 255;
    int l = idx >> 16;
    int s = l * DD * DD + k * DD + n;
    wt1[idx] = f2bf(w1[s]);
    wt2[idx] = f2bf(w2[s]);
}

// ---------------------------------------------------------------------------
// SpMM: P[b,n,:] = H[b,n,:] + sum_{j in CSR row n} H[b, cols[j], :]
// 64 lanes per row, 4 bf16 (8 bytes) per lane, f32 accumulation.
// ---------------------------------------------------------------------------
__global__ void __launch_bounds__(256)
spmm_kernel(const unsigned short* __restrict__ H,
            const int* __restrict__ rowPtr, const int* __restrict__ cols,
            unsigned short* __restrict__ P) {
    const int sub = threadIdx.x >> 6;
    const int ln  = threadIdx.x & 63;
    const int job = blockIdx.x * 4 + sub;       // == b*NN + n
    const int n   = job & (NN - 1);
    const int b   = job >> 11;
    const int d   = ln << 2;
    const size_t rowOff = (size_t)job * DD + d;

    uint2 u = *(const uint2*)(H + rowOff);
    float a0 = bflo(u.x), a1 = bfhi(u.x), a2 = bflo(u.y), a3 = bfhi(u.y);

    const unsigned short* Hb = H + ((size_t)b * NN) * DD + d;
    const int beg = rowPtr[n], end = rowPtr[n + 1];
    for (int j = beg; j < end; ++j) {
        const int c = cols[j];
        uint2 w = *(const uint2*)(Hb + (size_t)c * DD);
        a0 += bflo(w.x); a1 += bfhi(w.x); a2 += bflo(w.y); a3 += bfhi(w.y);
    }
    uint2 o;
    o.x = (unsigned int)f2bf(a0) | ((unsigned int)f2bf(a1) << 16);
    o.y = (unsigned int)f2bf(a2) | ((unsigned int)f2bf(a3) << 16);
    *(uint2*)(P + rowOff) = o;
}

// ---------------------------------------------------------------------------
// GEMM + bias + BN + ReLU, bf16 in / f32 accumulate / bf16 out.
// Block = 8 waves (256 threads), waves arranged 2 row-groups x 4 col-groups.
// Wave tile 64x32 (4x2 WMMA accumulators); block tile 128x128.
// K loop: 8 steps x (8 wmma / 6 fragment loads) -> 64 v_wmma per wave.
// ---------------------------------------------------------------------------
__global__ void __launch_bounds__(256)
gemm_bn_relu(const unsigned short* __restrict__ A,    // [ROWS, DD] bf16 row-major
             const unsigned short* __restrict__ Wt,   // [DD, DD]  bf16, Wt[n][k]
             const float* __restrict__ bias,
             const float* __restrict__ gamma, const float* __restrict__ beta,
             const float* __restrict__ mean,  const float* __restrict__ var,
             unsigned short* __restrict__ Out)        // [ROWS, DD] bf16
{
    const int lane   = threadIdx.x & 31;
    const int wave   = threadIdx.x >> 5;
    const int laneM  = lane & 15;
    const int laneHi = lane >> 4;
    const int rowBase = blockIdx.x * 128 + (wave & 1) * 64;   // 2 row-groups of 64
    const int colBase = blockIdx.y * 128 + (wave >> 1) * 32;  // 4 col-groups of 32

    v8f acc[4][2];
    #pragma unroll
    for (int mi = 0; mi < 4; ++mi)
        #pragma unroll
        for (int ni = 0; ni < 2; ++ni)
            acc[mi][ni] = (v8f){0.f, 0.f, 0.f, 0.f, 0.f, 0.f, 0.f, 0.f};

    const unsigned short* aRow = A  + (size_t)(rowBase + laneM) * DD;
    const unsigned short* bCol = Wt + (size_t)(colBase + laneM) * DD;
    const int kaOff = laneHi << 3;   // A lanes>=16 start at K+8
    const int kbOff = laneHi << 3;   // B lanes>=16 start at K+8

    #pragma unroll
    for (int k0 = 0; k0 < DD; k0 += 32) {
        Vec16 a[4], b[2];
        const int ka = k0 + kaOff;
        const int kb = k0 + kbOff;
        // A 16-bit 16x32 layout: halves 0..7 = K{ka..ka+7}, halves 8..15 = K{ka+16..ka+23}
        #pragma unroll
        for (int mi = 0; mi < 4; ++mi) {
            const unsigned short* p = aRow + (size_t)(mi * 16) * DD + ka;
            a[mi].q2[0] = *(const v4u*)(p);
            a[mi].q2[1] = *(const v4u*)(p + 16);
        }
        // B: lane = column, halves 0..7 = K{kb..kb+7}, halves 8..15 = K{kb+16..kb+23}
        #pragma unroll
        for (int ni = 0; ni < 2; ++ni) {
            const unsigned short* p = bCol + (size_t)(ni * 16) * DD + kb;
            b[ni].q2[0] = *(const v4u*)(p);
            b[ni].q2[1] = *(const v4u*)(p + 16);
        }
        #pragma unroll
        for (int mi = 0; mi < 4; ++mi)
            #pragma unroll
            for (int ni = 0; ni < 2; ++ni)
                acc[mi][ni] = __builtin_amdgcn_wmma_f32_16x16x32_bf16(
                    false, a[mi].h, false, b[ni].h, (short)0, acc[mi][ni], false, false);
    }

    // Epilogue: C/D layout — VGPR i: lanes0-15 M=i, lanes16-31 M=i+8; N = lane&15
    #pragma unroll
    for (int ni = 0; ni < 2; ++ni) {
        const int col = colBase + ni * 16 + laneM;
        const float s  = gamma[col] * rsqrtf(var[col] + 1e-5f);
        const float sh = beta[col] - mean[col] * s;
        const float bs = bias[col];
        #pragma unroll
        for (int mi = 0; mi < 4; ++mi) {
            #pragma unroll
            for (int i = 0; i < 8; ++i) {
                const int row = rowBase + mi * 16 + laneHi * 8 + i;
                float v = fmaxf((acc[mi][ni][i] + bs) * s + sh, 0.0f);
                Out[(size_t)row * DD + col] = f2bf(v);
            }
        }
    }
}

// ---------------------------------------------------------------------------
// score[b,q] += dot(H[b,q,:], pw) + pb[0]   (one block per (b,q))
// ---------------------------------------------------------------------------
__global__ void __launch_bounds__(256)
pred_accum(const unsigned short* __restrict__ H, const float* __restrict__ pw,
           const float* __restrict__ pb, float* __restrict__ score) {
    __shared__ float red[256];
    const int bq = blockIdx.x;           // < BB*QQ
    const int b = bq / QQ, q = bq - b * QQ;
    const int t = threadIdx.x;
    red[t] = bf2f(H[((size_t)(b * NN + q)) * DD + t]) * pw[t];
    __syncthreads();
    for (int s = 128; s > 0; s >>= 1) {
        if (t < s) red[t] += red[t + s];
        __syncthreads();
    }
    if (t == 0) score[bq] += red[0] + pb[0];
}

__global__ void zero_score(float* __restrict__ s) {
    int i = blockIdx.x * 256 + threadIdx.x;
    if (i < BB * QQ) s[i] = 0.0f;
}

__global__ void sigmoid_out(const float* __restrict__ score, float* __restrict__ out) {
    int i = blockIdx.x * 256 + threadIdx.x;
    if (i < BB * QQ) out[i] = 1.0f / (1.0f + expf(-score[i]));
}

// ---------------------------------------------------------------------------
extern "C" void kernel_launch(void* const* d_in, const int* in_sizes, int n_in,
                              void* d_out, int out_size, void* d_ws, size_t ws_size,
                              hipStream_t stream) {
    const float* batch = (const float*)d_in[0];
    const int*   edge  = (const int*)d_in[1];
    const float* emo   = (const float*)d_in[2];
    const float* w1    = (const float*)d_in[3];
    const float* b1    = (const float*)d_in[4];
    const float* mg    = (const float*)d_in[5];
    const float* mbt   = (const float*)d_in[6];
    const float* mmn   = (const float*)d_in[7];
    const float* mvr   = (const float*)d_in[8];
    const float* w2    = (const float*)d_in[9];
    const float* b2    = (const float*)d_in[10];
    const float* bg    = (const float*)d_in[11];
    const float* bbt   = (const float*)d_in[12];
    const float* bmn   = (const float*)d_in[13];
    const float* bvr   = (const float*)d_in[14];
    const float* pw    = (const float*)d_in[15];
    const float* pb    = (const float*)d_in[16];
    float* out = (float*)d_out;
    (void)in_sizes; (void)n_in; (void)out_size; (void)ws_size;

    const int* src = edge;        // edge_mat[0]
    const int* dst = edge + EE;   // edge_mat[1]

    char* base = (char*)d_ws;
    size_t off = 0;
    auto take = [&](size_t bytes) -> char* {
        char* p = base + off;
        off += (bytes + 255) & ~(size_t)255;
        return p;
    };
    unsigned short* H   = (unsigned short*)take((size_t)ROWS * DD * 2);  // 32 MB
    unsigned short* P   = (unsigned short*)take((size_t)ROWS * DD * 2);  // 32 MB
    unsigned short* T   = (unsigned short*)take((size_t)ROWS * DD * 2);  // 32 MB
    unsigned short* Wt1 = (unsigned short*)take((size_t)LL * DD * DD * 2);
    unsigned short* Wt2 = (unsigned short*)take((size_t)LL * DD * DD * 2);
    float* score = (float*)take((size_t)BB * QQ * 4);
    int* counts  = (int*)take((size_t)NN * 4);
    int* rowPtr  = (int*)take((size_t)(NN + 1) * 4);
    int* colsBuf = (int*)take((size_t)EE * 4);

    zero_score<<<2, 256, 0, stream>>>(score);
    csr_count<<<(NN + 255) / 256, 256, 0, stream>>>(src, counts);
    csr_scan<<<1, 1, 0, stream>>>(counts, rowPtr);
    csr_fill<<<(NN + 255) / 256, 256, 0, stream>>>(src, dst, rowPtr, colsBuf);
    init_h<<<ROWS, 256, 0, stream>>>(batch, emo, H);            // ROWS*DD/256 blocks
    conv_w<<<(LL * DD * DD) / 256, 256, 0, stream>>>(w1, w2, Wt1, Wt2);

    pred_accum<<<BB * QQ, 256, 0, stream>>>(H, pw, pb, score);  // hidden[0]

    for (int l = 0; l < LL; ++l) {
        spmm_kernel<<<ROWS / 4, 256, 0, stream>>>(H, rowPtr, colsBuf, P);
        gemm_bn_relu<<<dim3(ROWS / 128, DD / 128), 256, 0, stream>>>(
            P, Wt1 + l * DD * DD, b1 + l * DD,
            mg + l * DD, mbt + l * DD, mmn + l * DD, mvr + l * DD, T);
        gemm_bn_relu<<<dim3(ROWS / 128, DD / 128), 256, 0, stream>>>(
            T, Wt2 + l * DD * DD, b2 + l * DD,
            bg + l * DD, bbt + l * DD, bmn + l * DD, bvr + l * DD, H);
        pred_accum<<<BB * QQ, 256, 0, stream>>>(H, pw + (l + 1) * DD, pb + (l + 1), score);
    }
    sigmoid_out<<<2, 256, 0, stream>>>(score, out);
}
